// TransformerBlock_21723944583721
// MI455X (gfx1250) — matrix operand order repaired and checked by reading, not compiled, not run on hardware
//
#include <hip/hip_runtime.h>

#define T_TOK 2048
#define HDIM  1024
#define FDIM  2048
#define NHEADS 16
#define HEADD  64
#define NEXP   8

typedef __attribute__((ext_vector_type(16))) __bf16 v16bf;
typedef __attribute__((ext_vector_type(8)))  float  v8f;
typedef __attribute__((ext_vector_type(4)))  int    v4i;

#define AS1 __attribute__((address_space(1)))
#define AS3 __attribute__((address_space(3)))

#if __has_builtin(__builtin_amdgcn_global_load_async_to_lds_b128) && \
    __has_builtin(__builtin_amdgcn_s_wait_asynccnt)
#define HAVE_ASYNC_LDS 1
#else
#define HAVE_ASYNC_LDS 0
#endif

__device__ __forceinline__ unsigned short f2bf(float f) {
  unsigned int u = __float_as_uint(f);
  unsigned int r = u + 0x7FFFu + ((u >> 16) & 1u);
  return (unsigned short)(r >> 16);
}
__device__ __forceinline__ float bf2f(unsigned short h) {
  return __uint_as_float(((unsigned int)h) << 16);
}
__device__ __forceinline__ v8f zero8() {
  v8f z = {0.f,0.f,0.f,0.f,0.f,0.f,0.f,0.f};
  return z;
}

union ABf    { v16bf v; uint4 q[2]; };
union Pack8  { unsigned short s[8];  uint4 q; };

__device__ __forceinline__ v8f wmma_bf16(v16bf a, v16bf b, v8f c) {
  // D = A(16x32 bf16) * B(32x16 bf16) + C(16x16 f32)
  return __builtin_amdgcn_wmma_f32_16x16x32_bf16(false, a, false, b, (short)0, c,
                                                 false, false);
}

// ---------------------------------------------------------------------------
// LayerNorm: x[T,1024] -> bf16 out; optionally copy x to copy_out (residual init)
// ---------------------------------------------------------------------------
__global__ __launch_bounds__(256) void ln_k(const float* __restrict__ x,
                                            const float* __restrict__ w,
                                            const float* __restrict__ b,
                                            unsigned short* __restrict__ outb,
                                            float* __restrict__ copy_out) {
  int t = blockIdx.x;
  const float* row = x + (size_t)t * HDIM;
  int tid = threadIdx.x, lane = tid & 31, wave = tid >> 5;
  float vals[4], s = 0.f, s2 = 0.f;
#pragma unroll
  for (int i = 0; i < 4; i++) {
    float v = row[tid + i * 256];
    vals[i] = v; s += v; s2 += v * v;
  }
#pragma unroll
  for (int o = 16; o; o >>= 1) {
    s  += __shfl_xor(s,  o, 32);
    s2 += __shfl_xor(s2, o, 32);
  }
  __shared__ float sh[16];
  if (lane == 0) { sh[wave] = s; sh[8 + wave] = s2; }
  __syncthreads();
  s = 0.f; s2 = 0.f;
#pragma unroll
  for (int i = 0; i < 8; i++) { s += sh[i]; s2 += sh[8 + i]; }
  float mu = s * (1.0f / HDIM);
  float var = s2 * (1.0f / HDIM) - mu * mu;
  float rs = rsqrtf(var + 1e-5f);
#pragma unroll
  for (int i = 0; i < 4; i++) {
    int idx = tid + i * 256;
    float hv = (vals[i] - mu) * rs * w[idx] + b[idx];
    outb[(size_t)t * HDIM + idx] = f2bf(hv);
    if (copy_out) copy_out[(size_t)t * HDIM + idx] = vals[i];
  }
}

// ---------------------------------------------------------------------------
// GEMM: C[M,N] = A(bf16)[M,Kd] * op(B fp32) (+ epilogue)
//  BT=true : B is [N,Kd] row-major (use B^T)   ldb == Kd
//  BT=false: B is [Kd,N] row-major             ldb == N
//  EPI 0: +bias(col)           -> bf16
//  EPI 1: +bias(col)+res[M,N]  -> f32
//  EPI 2: gelu(x)*gate(row*8)  -> bf16
//  EPI 3: accumulate           -> f32 +=
// Block tile 128x64, 8 waves (16 rows each x 64 cols), K step 32.
// A tile is staged global->LDS with CDNA5 async-to-LDS copies when available.
// ---------------------------------------------------------------------------
#define APAD 40
#define BPAD 40

template <int EPI, bool BT>
__global__ __launch_bounds__(256) void gemm_bf16_k(
    const unsigned short* __restrict__ A, const float* __restrict__ B,
    void* __restrict__ Cout, int M, int N, int Kd, int ldb,
    const float* __restrict__ ep0, const float* __restrict__ ep1) {
  (void)M;
  __shared__ unsigned short sA[128 * APAD];
  __shared__ unsigned short sB[64 * BPAD];
  int tid = threadIdx.x;
  int lane = tid & 31, wave = tid >> 5;
  int c = lane >> 4, mm = lane & 15;
  int m0 = blockIdx.y * 128;
  int n0 = blockIdx.x * 64;

  v8f acc[4] = {zero8(), zero8(), zero8(), zero8()};

  for (int k0 = 0; k0 < Kd; k0 += 32) {
    // ---- stage A tile 128x32 bf16: global -> LDS (async when available) ----
    {
      int row = tid >> 1;
      int kc  = (tid & 1) * 16;
      const unsigned short* gsrc = A + (size_t)(m0 + row) * Kd + k0 + kc;
      unsigned short* ldst = sA + row * APAD + kc;
#if HAVE_ASYNC_LDS
      __builtin_amdgcn_global_load_async_to_lds_b128((AS1 v4i*)gsrc,
                                                     (AS3 v4i*)ldst, 0, 0);
      __builtin_amdgcn_global_load_async_to_lds_b128((AS1 v4i*)(gsrc + 8),
                                                     (AS3 v4i*)(ldst + 8), 0, 0);
#else
      const uint4* src = (const uint4*)gsrc;
      uint4 d0 = src[0], d1 = src[1];
      *(uint4*)ldst       = d0;
      *(uint4*)(ldst + 8) = d1;
#endif
    }
    // ---- stage B tile 32k x 64n, fp32 -> bf16, stored [n][k] ----
    {
      int col = tid & 63;
      int kc  = (tid >> 6) * 8;
      float f[8];
      if (BT) {
        const float* src = B + (size_t)(n0 + col) * ldb + k0 + kc;
#pragma unroll
        for (int j = 0; j < 8; j++) f[j] = src[j];
      } else {
        const float* src = B + (size_t)(k0 + kc) * ldb + n0 + col;
#pragma unroll
        for (int j = 0; j < 8; j++) f[j] = src[(size_t)j * ldb];
      }
      Pack8 pk;
#pragma unroll
      for (int j = 0; j < 8; j++) pk.s[j] = f2bf(f[j]);
      *(uint4*)(sB + col * BPAD + kc) = pk.q;
    }
#if HAVE_ASYNC_LDS
    __builtin_amdgcn_s_wait_asynccnt(0);
#endif
    __syncthreads();

    // ---- compute: wave owns rows [m0 + wave*16, +16), cols [n0, n0+64) ----
    ABf afr;
    afr.q[0] = *(const uint4*)(sA + (wave * 16 + mm) * APAD + c * 8);
    afr.q[1] = *(const uint4*)(sA + (wave * 16 + mm) * APAD + 16 + c * 8);
#pragma unroll
    for (int nn = 0; nn < 4; nn++) {
      ABf bfr;
      bfr.q[0] = *(const uint4*)(sB + (nn * 16 + mm) * BPAD + c * 8);
      bfr.q[1] = *(const uint4*)(sB + (nn * 16 + mm) * BPAD + 16 + c * 8);
      acc[nn] = wmma_bf16(afr.v, bfr.v, acc[nn]);
    }
    __syncthreads();
  }

  // ---- epilogue ----
  int gm_base = m0 + wave * 16;
#pragma unroll
  for (int nn = 0; nn < 4; nn++) {
#pragma unroll
    for (int r = 0; r < 8; r++) {
      int gm = gm_base + r + 8 * c;
      int gn = n0 + nn * 16 + mm;
      float v = acc[nn][r];
      if (EPI == 0) {
        v += ep0[gn];
        ((unsigned short*)Cout)[(size_t)gm * N + gn] = f2bf(v);
      } else if (EPI == 1) {
        v += ep0[gn] + ep1[(size_t)gm * N + gn];
        ((float*)Cout)[(size_t)gm * N + gn] = v;
      } else if (EPI == 2) {
        float g = 0.5f * v * (1.0f + erff(v * 0.70710678f));
        g *= ep0[(size_t)gm * 8];  // ep0 = gates + expert, row stride 8
        ((unsigned short*)Cout)[(size_t)gm * N + gn] = f2bf(g);
      } else {
        float* o = (float*)Cout + (size_t)gm * N + gn;
        *o += v;
      }
    }
  }
}

// ---------------------------------------------------------------------------
// Flash attention: one wave per (16-query tile, head). qkv bf16 [T, 3H].
// Scores = Q K^T / 8 with causal mask; online softmax; O = P V.
// V tile staged coalesced through LDS with transpose-at-store.
// ---------------------------------------------------------------------------
__global__ __launch_bounds__(32) void attn_k(const unsigned short* __restrict__ qkv,
                                             unsigned short* __restrict__ outb) {
  const int TH = 3 * HDIM;
  int head = blockIdx.y;
  int q0 = blockIdx.x * 16;
  int lane = threadIdx.x & 31;
  int c = lane >> 4, mm = lane & 15;

  __shared__ unsigned short sp[16 * 40];  // P transpose staging (bf16)
  __shared__ unsigned short sV[64 * 40];  // V^T tile: [d][key], key stride 40

  // Q A-fragments, two K-steps of 32 over head dim 64
  ABf aq[2];
  {
    const unsigned short* qrow = qkv + (size_t)(q0 + mm) * TH + head * HEADD;
#pragma unroll
    for (int ks = 0; ks < 2; ks++) {
      aq[ks].q[0] = *(const uint4*)(qrow + ks * 32 + c * 8);
      aq[ks].q[1] = *(const uint4*)(qrow + ks * 32 + 16 + c * 8);
    }
  }

  v8f o[4] = {zero8(), zero8(), zero8(), zero8()};
  float mrow[8], lrow[8];
#pragma unroll
  for (int r = 0; r < 8; r++) { mrow[r] = -1e30f; lrow[r] = 0.f; }

  for (int t0 = 0; t0 <= q0; t0 += 32) {
    // ---- stage V tile (32 keys x 64 d) transposed: sV[d][key] ----
    // lane = key row; coalesced 128B per lane in, b16 scatter to LDS.
    {
      const unsigned short* vrow =
          qkv + (size_t)(t0 + lane) * TH + 2 * HDIM + head * HEADD;
#pragma unroll
      for (int i = 0; i < 8; i++) {
        uint4 d4 = *(const uint4*)(vrow + i * 8);
        unsigned int wds[4] = {d4.x, d4.y, d4.z, d4.w};
#pragma unroll
        for (int j = 0; j < 4; j++) {
          sV[(i * 8 + j * 2 + 0) * 40 + lane] = (unsigned short)(wds[j] & 0xFFFFu);
          sV[(i * 8 + j * 2 + 1) * 40 + lane] = (unsigned short)(wds[j] >> 16);
        }
      }
    }

    // ---- scores S = Q K^T ----
    v8f sc[2];
#pragma unroll
    for (int half = 0; half < 2; half++) {
      int key = t0 + half * 16 + mm;  // lane's column = key index
      const unsigned short* krow = qkv + (size_t)key * TH + HDIM + head * HEADD;
      ABf bk0, bk1;
      bk0.q[0] = *(const uint4*)(krow + c * 8);
      bk0.q[1] = *(const uint4*)(krow + 16 + c * 8);
      bk1.q[0] = *(const uint4*)(krow + 32 + c * 8);
      bk1.q[1] = *(const uint4*)(krow + 32 + 16 + c * 8);
      v8f t = zero8();
      t = wmma_bf16(aq[0].v, bk0.v, t);
      t = wmma_bf16(aq[1].v, bk1.v, t);
#pragma unroll
      for (int r = 0; r < 8; r++) {
        int gq = q0 + r + 8 * c;
        int gk = t0 + half * 16 + mm;
        float sv = t[r] * 0.125f;  // 1/sqrt(64)
        if (gk > gq) sv = -1e30f;
        t[r] = sv;
      }
      sc[half] = t;
    }

    // ---- online softmax per query row ----
#pragma unroll
    for (int r = 0; r < 8; r++) {
      float rm = fmaxf(sc[0][r], sc[1][r]);
#pragma unroll
      for (int off = 8; off; off >>= 1) rm = fmaxf(rm, __shfl_xor(rm, off, 32));
      float mn = fmaxf(mrow[r], rm);
      float scale = __expf(mrow[r] - mn);
      float p0 = __expf(sc[0][r] - mn);
      float p1 = __expf(sc[1][r] - mn);
      float ps = p0 + p1;
#pragma unroll
      for (int off = 8; off; off >>= 1) ps += __shfl_xor(ps, off, 32);
      lrow[r] = lrow[r] * scale + ps;
      mrow[r] = mn;
      o[0][r] *= scale; o[1][r] *= scale; o[2][r] *= scale; o[3][r] *= scale;
      // stage P (C-layout) -> LDS as [query][key] bf16
      sp[(r + 8 * c) * 40 + mm]      = f2bf(p0);
      sp[(r + 8 * c) * 40 + 16 + mm] = f2bf(p1);
    }

    // reload P as A-fragment (16 queries x 32 keys)
    ABf ap;
    ap.q[0] = *(const uint4*)(sp + mm * 40 + c * 8);
    ap.q[1] = *(const uint4*)(sp + mm * 40 + 16 + c * 8);

    // ---- O += P V : B-fragments from transposed LDS tile ----
#pragma unroll
    for (int nn = 0; nn < 4; nn++) {
      int d = nn * 16 + mm;  // lane's column = head-dim index
      ABf bv;
      bv.q[0] = *(const uint4*)(sV + d * 40 + c * 8);
      bv.q[1] = *(const uint4*)(sV + d * 40 + 16 + c * 8);
      o[nn] = wmma_bf16(ap.v, bv.v, o[nn]);
    }
  }

  // normalize + store bf16 [T, H]
#pragma unroll
  for (int nn = 0; nn < 4; nn++) {
#pragma unroll
    for (int r = 0; r < 8; r++) {
      float ov = o[nn][r] / lrow[r];
      outb[(size_t)(q0 + r + 8 * c) * HDIM + head * HEADD + nn * 16 + mm] = f2bf(ov);
    }
  }
}

// ---------------------------------------------------------------------------
// Router: logits = hn2 @ rw^T ; softmax over 8; top-2 -> gates[T,8]
// One wave per token.
// ---------------------------------------------------------------------------
__global__ __launch_bounds__(256) void router_k(const unsigned short* __restrict__ hn2,
                                                const float* __restrict__ rw,
                                                float* __restrict__ gates) {
  int t = blockIdx.x * 8 + (threadIdx.x >> 5);
  int lane = threadIdx.x & 31;
  float logit[NEXP];
#pragma unroll
  for (int e = 0; e < NEXP; e++) {
    float p = 0.f;
    for (int h = lane; h < HDIM; h += 32)
      p += bf2f(hn2[(size_t)t * HDIM + h]) * rw[(size_t)e * HDIM + h];
#pragma unroll
    for (int o = 16; o; o >>= 1) p += __shfl_xor(p, o, 32);
    logit[e] = p;
  }
  if (lane == 0) {
    float mx = logit[0];
#pragma unroll
    for (int e = 1; e < NEXP; e++) mx = fmaxf(mx, logit[e]);
    float pr[NEXP], den = 0.f;
#pragma unroll
    for (int e = 0; e < NEXP; e++) { pr[e] = __expf(logit[e] - mx); den += pr[e]; }
#pragma unroll
    for (int e = 0; e < NEXP; e++) pr[e] /= den;
    int i1 = 0;
#pragma unroll
    for (int e = 1; e < NEXP; e++) if (pr[e] > pr[i1]) i1 = e;
    int i2 = -1;
#pragma unroll
    for (int e = 0; e < NEXP; e++)
      if (e != i1 && (i2 < 0 || pr[e] > pr[i2])) i2 = e;
#pragma unroll
    for (int e = 0; e < NEXP; e++)
      gates[(size_t)t * 8 + e] = (e == i1 || e == i2) ? pr[e] : 0.f;
  }
}

// ---------------------------------------------------------------------------
extern "C" void kernel_launch(void* const* d_in, const int* in_sizes, int n_in,
                              void* d_out, int out_size, void* d_ws, size_t ws_size,
                              hipStream_t stream) {
  (void)in_sizes; (void)n_in; (void)out_size; (void)ws_size;
  const float* x    = (const float*)d_in[0];
  const float* ln1w = (const float*)d_in[1];
  const float* ln1b = (const float*)d_in[2];
  const float* inw  = (const float*)d_in[3];   // [3H, H]
  const float* inb  = (const float*)d_in[4];
  const float* outw = (const float*)d_in[5];   // [H, H]
  const float* outb = (const float*)d_in[6];
  const float* ln2w = (const float*)d_in[7];
  const float* ln2b = (const float*)d_in[8];
  const float* rw   = (const float*)d_in[9];   // [E, H]
  const float* w1   = (const float*)d_in[10];  // [E, H, F]
  const float* w2   = (const float*)d_in[11];  // [E, F, H]
  float* out = (float*)d_out;

  char* ws = (char*)d_ws;
  unsigned short* hn   = (unsigned short*)(ws);                        //  4 MB
  unsigned short* qkv  = (unsigned short*)(ws + ((size_t)4  << 20));   // 12 MB
  unsigned short* attn = (unsigned short*)(ws + ((size_t)16 << 20));   //  4 MB
  float*          x1   = (float*)         (ws + ((size_t)20 << 20));   //  8 MB
  unsigned short* hn2  = (unsigned short*)(ws + ((size_t)28 << 20));   //  4 MB
  float*          gate = (float*)         (ws + ((size_t)32 << 20));   // 64 KB
  unsigned short* hmid = (unsigned short*)(ws + ((size_t)33 << 20));   //  8 MB

  // 1) LN1 -> hn (bf16)
  ln_k<<<T_TOK, 256, 0, stream>>>(x, ln1w, ln1b, hn, nullptr);
  // 2) qkv = hn @ inw^T + inb   [2048, 3072]
  gemm_bf16_k<0, true><<<dim3(3 * HDIM / 64, T_TOK / 128), 256, 0, stream>>>(
      hn, inw, qkv, T_TOK, 3 * HDIM, HDIM, HDIM, inb, nullptr);
  // 3) causal flash attention per head
  attn_k<<<dim3(T_TOK / 16, NHEADS), 32, 0, stream>>>(qkv, attn);
  // 4) x1 = attn @ outw^T + outb + x   [2048, 1024] fp32
  gemm_bf16_k<1, true><<<dim3(HDIM / 64, T_TOK / 128), 256, 0, stream>>>(
      attn, outw, x1, T_TOK, HDIM, HDIM, HDIM, outb, x);
  // 5) LN2 -> hn2 (bf16); also d_out = x1 (residual base)
  ln_k<<<T_TOK, 256, 0, stream>>>(x1, ln2w, ln2b, hn2, out);
  // 6) router gates
  router_k<<<T_TOK / 8, 256, 0, stream>>>(hn2, rw, gate);
  // 7) experts: hmid = gelu(hn2 @ w1_e) * gate_e ; d_out += hmid @ w2_e
  for (int e = 0; e < NEXP; e++) {
    const float* w1e = w1 + (size_t)e * HDIM * FDIM;  // [H, F] = [K, N]
    const float* w2e = w2 + (size_t)e * FDIM * HDIM;  // [F, H] = [K, N]
    gemm_bf16_k<2, false><<<dim3(FDIM / 64, T_TOK / 128), 256, 0, stream>>>(
        hn2, w1e, hmid, T_TOK, FDIM, HDIM, FDIM, gate + e, nullptr);
    gemm_bf16_k<3, false><<<dim3(HDIM / 64, T_TOK / 128), 256, 0, stream>>>(
        hmid, w2e, out, T_TOK, HDIM, FDIM, HDIM, nullptr, nullptr);
  }
}
